// MultiHeadAttentionLayer_70136815944233
// MI455X (gfx1250) — matrix operand order repaired and checked by reading, not compiled, not run on hardware
//
#include <hip/hip_runtime.h>
#include <hip/hip_bf16.h>
#include <math.h>

// ---------------------------------------------------------------------------
// Graph attention layer, algebraically collapsed:
//   weights_final[i,j,h] = 0                         (i==j)
//                        = exp(clip(w_real))/1.1     (winning real edge i->j)
//                        = 0.1/1.1 (=c0)             (otherwise)
//   out[j] = (c0*(sumV - V[j]) + sum_edges Δ*V[i]) / (c0*(N-1) + sum Δ + 1e-6)
// The "fake" branch (WQ2/WK2/WE2) is provably constant off-edge and
// overwritten on-edge, so it never touches the output.
//
// GEMMs (h@W, e@WE) run on v_wmma_f32_16x16x32_f16 (f16 A/B, f32 accum).
// ---------------------------------------------------------------------------

typedef __attribute__((ext_vector_type(16))) _Float16 v16h;
typedef __attribute__((ext_vector_type(8)))  float    v8f;

#define NN     1024
#define EDGES  16384

// C[M x 64] = A[M x 64] * B[64 x 64], M multiple of 32.
// One wave per 16x16 output tile; 8 waves/block cover 2 M-tiles x 4 N-tiles.
__global__ void __launch_bounds__(256) gemm64_wmma(const float* __restrict__ A,
                                                   const float* __restrict__ B,
                                                   float* __restrict__ C) {
  const int wave = threadIdx.x >> 5;
  const int lane = threadIdx.x & 31;
  const int mt   = blockIdx.x * 2 + (wave >> 2);
  const int nt   = wave & 3;
  const int half = lane >> 4;   // lane group 0-15 / 16-31
  const int l15  = lane & 15;
  const int arow = mt * 16 + l15;   // A: lane carries one M row
  const int bcol = nt * 16 + l15;   // B/D: lane carries one N column

  v8f acc = {};
  #pragma unroll
  for (int kt = 0; kt < 2; ++kt) {
    const int kb = kt * 32;
    v16h a, b;
    // A 16x32 f16 fragment: lanes 0-15 hold K 0..15, lanes 16-31 hold K 16..31
    #pragma unroll
    for (int g = 0; g < 8; ++g) {
      const int k = kb + half * 16 + 2 * g;
      a[2 * g]     = (_Float16)A[arow * 64 + k];
      a[2 * g + 1] = (_Float16)A[arow * 64 + k + 1];
    }
    // B 32x16 f16 fragment: lanes 0-15 hold K 0..15, lanes 16-31 hold K 16..31
    #pragma unroll
    for (int g = 0; g < 8; ++g) {
      const int k = kb + half * 16 + 2 * g;
      b[2 * g]     = (_Float16)B[k * 64 + bcol];
      b[2 * g + 1] = (_Float16)B[(k + 1) * 64 + bcol];
    }
    acc = __builtin_amdgcn_wmma_f32_16x16x32_f16(false, a, false, b,
                                                 (short)0, acc, false, false);
  }
  // D 16x16 f32: VGPR g holds row (g + 8*half), lane l15 = column
  #pragma unroll
  for (int g = 0; g < 8; ++g) {
    C[(mt * 16 + half * 8 + g) * 64 + bcol] = acc[g];
  }
}

__global__ void zero_f32(float* __restrict__ p, int n) {
  int i = blockIdx.x * blockDim.x + threadIdx.x;
  if (i < n) p[i] = 0.0f;
}

__global__ void fill_i32(int* __restrict__ p, int n, int v) {
  int i = blockIdx.x * blockDim.x + threadIdx.x;
  if (i < n) p[i] = v;
}

// Duplicate-(src,dst) resolution: highest edge index wins (last-write).
__global__ void dedup_max(const int* __restrict__ src, const int* __restrict__ dst,
                          int* __restrict__ win) {
  int k = blockIdx.x * blockDim.x + threadIdx.x;
  if (k < EDGES) atomicMax(&win[src[k] * NN + dst[k]], k);
}

// One thread per (edge, head): triple-product score, exp, scatter-add to dst.
__global__ void edge_attn(const float* __restrict__ Qh, const float* __restrict__ Kh,
                          const float* __restrict__ Vh, const float* __restrict__ Ee,
                          const int* __restrict__ src, const int* __restrict__ dst,
                          const int* __restrict__ win,
                          float* __restrict__ acc_out, float* __restrict__ acc_z) {
  int t = blockIdx.x * blockDim.x + threadIdx.x;
  if (t >= EDGES * 8) return;
  const int k  = t >> 3;
  const int hh = t & 7;
  const int i = src[k];
  const int j = dst[k];
  if (i == j) return;                        // diagonal zeroed anyway
  if (win && win[i * NN + j] != k) return;   // duplicate lost to later edge

  const float* Kr = Kh + i * 64 + hh * 8;
  const float* Qr = Qh + j * 64 + hh * 8;
  const float* Er = Ee + k * 64 + hh * 8;
  float w = 0.0f;
  #pragma unroll
  for (int d = 0; d < 8; ++d) w += Kr[d] * Qr[d] * Er[d];
  w *= 0.3535533905932738f;                  // 1/sqrt(D), D=8
  w = fminf(5.0f, fmaxf(-5.0f, w));
  const float delta = (expf(w) - 0.1f) * (1.0f / 1.1f);  // exp/1.1 - c0

  atomicAdd(&acc_z[j * 8 + hh], delta);
  const float* Vr = Vh + i * 64 + hh * 8;
  float* Or = acc_out + j * 64 + hh * 8;
  #pragma unroll
  for (int d = 0; d < 8; ++d) atomicAdd(&Or[d], delta * Vr[d]);
}

// Column sums of Vh: SV[c] = sum_rows Vh[r][c]  (one block, 64 lanes, coalesced)
__global__ void colsum64(const float* __restrict__ Vh, float* __restrict__ SV) {
  int c = threadIdx.x;
  float s = 0.0f;
  for (int r = 0; r < NN; ++r) s += Vh[r * 64 + c];
  SV[c] = s;
}

__global__ void finalize(const float* __restrict__ Vh, const float* __restrict__ SV,
                         const float* __restrict__ acc_out, const float* __restrict__ acc_z,
                         float* __restrict__ out) {
  int t = blockIdx.x * blockDim.x + threadIdx.x;
  if (t >= NN * 64) return;
  const int j  = t >> 6;
  const int c  = t & 63;
  const int hh = c >> 3;
  const float c0  = 0.1f / 1.1f;
  const float num = c0 * (SV[c] - Vh[t]) + acc_out[t];
  const float den = c0 * (float)(NN - 1) + acc_z[j * 8 + hh] + 1e-6f;
  out[t] = num / den;
}

extern "C" void kernel_launch(void* const* d_in, const int* in_sizes, int n_in,
                              void* d_out, int out_size, void* d_ws, size_t ws_size,
                              hipStream_t stream) {
  const float* h   = (const float*)d_in[0];
  const float* e   = (const float*)d_in[1];
  const float* WQ  = (const float*)d_in[2];
  const float* WK  = (const float*)d_in[3];
  const float* WE  = (const float*)d_in[4];
  // d_in[5..7] = WQ2, WK2, WE2: provably unused (fake branch is constant off-edge
  // and overwritten on-edge).
  const float* WV  = (const float*)d_in[8];
  const int* src   = (const int*)d_in[9];
  const int* dst   = (const int*)d_in[10];
  float* out       = (float*)d_out;

  // Workspace layout (bytes, 256-aligned chunks)
  char* ws = (char*)d_ws;
  float* Qh      = (float*)(ws + 0);                    // 1024*64 f32 = 256KB
  float* Kh      = (float*)(ws + 262144);               // 256KB
  float* Vh      = (float*)(ws + 524288);               // 256KB
  float* acc_out = (float*)(ws + 786432);               // 256KB
  float* acc_z   = (float*)(ws + 1048576);              // 32KB
  float* SV      = (float*)(ws + 1081344);              // 256B
  float* Ee      = (float*)(ws + 1081600);              // 16384*64 f32 = 4MB
  int*   winner  = (int*)  (ws + 5275904);              // 1024*1024 i32 = 4MB
  const size_t need_full = 5275904 + (size_t)NN * NN * 4;
  if (ws_size < need_full) winner = nullptr;            // fall back: no dedup

  // 1) zero accumulators (acc_out + acc_z contiguous = 73728 floats)
  zero_f32<<<dim3((73728 + 255) / 256), dim3(256), 0, stream>>>(acc_out, 73728);
  // 2) reset winner map each call (harness does not re-poison ws)
  if (winner)
    fill_i32<<<dim3((NN * NN + 255) / 256), dim3(256), 0, stream>>>(winner, NN * NN, -1);

  // 3) WMMA projections: Qh/Kh/Vh = h @ W*, Ee = e @ WE
  gemm64_wmma<<<dim3(NN / 32),    dim3(256), 0, stream>>>(h, WQ, Qh);
  gemm64_wmma<<<dim3(NN / 32),    dim3(256), 0, stream>>>(h, WK, Kh);
  gemm64_wmma<<<dim3(NN / 32),    dim3(256), 0, stream>>>(h, WV, Vh);
  gemm64_wmma<<<dim3(EDGES / 32), dim3(256), 0, stream>>>(e, WE, Ee);

  // 4) duplicate-edge resolution (last write wins)
  if (winner)
    dedup_max<<<dim3((EDGES + 255) / 256), dim3(256), 0, stream>>>(src, dst, winner);

  // 5) per-(edge,head) attention correction, scattered by destination
  edge_attn<<<dim3((EDGES * 8 + 255) / 256), dim3(256), 0, stream>>>(
      Qh, Kh, Vh, Ee, src, dst, winner, acc_out, acc_z);

  // 6) value column sums for the constant-background term
  colsum64<<<dim3(1), dim3(64), 0, stream>>>(Vh, SV);

  // 7) combine background + corrections, normalize
  finalize<<<dim3((NN * 64 + 255) / 256), dim3(256), 0, stream>>>(
      Vh, SV, acc_out, acc_z, out);
}
